// AllInOneFlow_14757507629669
// MI455X (gfx1250) — matrix operand order repaired and checked by reading, not compiled, not run on hardware
//
#include <hip/hip_runtime.h>

#define B_TOTAL 131072
#define D 64
#define D1 32
#define D2 32
#define H 256

#define WAVES 8
#define ROWS_PER_WAVE 16
#define ROWS_PER_BLOCK (WAVES * ROWS_PER_WAVE)   // 128
#define TILES_PER_BLOCK 2
#define NBLOCKS (B_TOTAL / (ROWS_PER_BLOCK * TILES_PER_BLOCK))   // 512

// padded LDS row strides (conflict-free bank patterns for b128 fragment loads)
#define W1S 40     // halfs per row of W1t  (K=32 used)
#define WS  264    // halfs per row of W2t / W3t / h1 / h2 (K=256 used)
#define YS  68     // floats per row of y staging

typedef _Float16 v16h __attribute__((ext_vector_type(16)));
typedef _Float16 v8h  __attribute__((ext_vector_type(8)));
typedef float    v8f  __attribute__((ext_vector_type(8)));

// branch-free fast tanh: copysign(1 - 2*rcp(1+e^{2|x|}), x); exp saturation-safe
__device__ __forceinline__ float tanh_fast(float x) {
    float ax = __builtin_fabsf(x);
    float e  = __expf(ax + ax);
    float r  = __builtin_amdgcn_rcpf(e + 1.0f);
    float t  = __builtin_fmaf(-2.0f, r, 1.0f);
    return __builtin_copysignf(t, x);
}

// tanh-approx gelu rewritten as x * sigmoid(2*k0*(x + k1 x^3)); branch-free
__device__ __forceinline__ float gelu_fast(float x) {
    const float c2  = 1.5957691216057308f;   // 2*sqrt(2/pi)
    const float k1  = 0.044715f;
    float x2 = x * x;
    float u  = x * __builtin_fmaf(c2 * k1, x2, c2);   // 2*k0*(x + k1 x^3)
    float s  = __expf(-u);
    float r  = __builtin_amdgcn_rcpf(1.0f + s);
    return x * r;
}

__device__ __forceinline__ v8f bcast8(float v) {
    v8f r = {v, v, v, v, v, v, v, v};
    return r;
}

__device__ __forceinline__ v16h join8(v8h lo, v8h hi) {
    return __builtin_shufflevector(lo, hi, 0,1,2,3,4,5,6,7,8,9,10,11,12,13,14,15);
}

union WaveScratch {
    _Float16 h[ROWS_PER_WAVE * WS];    // 8448 B (layer-1 activations)
    float    y[ROWS_PER_WAVE * YS];    // 4352 B (output staging, aliases h)
};

struct SMem {
    _Float16 W1t[H * W1S];             //  20480 B  [n][k] padded
    _Float16 W2t[H * WS];              // 135168 B  [n][k] padded
    _Float16 W3t[D * WS];              //  33792 B  [n][k] padded
    float    B1[H];
    float    B2[H];
    float    B3[D];
    float    Egls[D];                  // exp(4*tanh(g/4))
    float    Goff[D];
    float    SumGlsPart[2];
    unsigned char InvPerm[D];
    WaveScratch scrA[WAVES];           // 67584 B
    _Float16 h2[WAVES][ROWS_PER_WAVE * WS];  // 67584 B
};  // 327,496 B  (< 320 KiB WGP limit)

__global__ __launch_bounds__(256, 1)
void flow_fused_kernel(const float* __restrict__ x,
                       const int*   __restrict__ perm,
                       const float* __restrict__ W1, const float* __restrict__ b1,
                       const float* __restrict__ W2, const float* __restrict__ b2,
                       const float* __restrict__ W3, const float* __restrict__ b3,
                       const float* __restrict__ g_ls, const float* __restrict__ g_off,
                       float* __restrict__ out_y, float* __restrict__ out_lj)
{
    __shared__ SMem sm;

    const int tid    = threadIdx.x;
    const int lane   = tid & 31;
    const int wave   = tid >> 5;
    const int mrow   = lane & 15;
    const int hiHalf = lane >> 4;

    // ---------------- phase 0: stage weights / constants in LDS (once) --------
    if (tid < D) {
        int p = perm[tid];
        sm.InvPerm[p] = (unsigned char)tid;        // inv_perm = argsort(perm)
        float g  = g_ls[tid];
        float gl = 4.0f * tanh_fast(0.25f * g);
        sm.Egls[tid] = __expf(gl);
        sm.Goff[tid] = g_off[tid];
        sm.B3[tid]   = b3[tid];
        float s = gl;                               // wave32 reduction of gls
        s += __shfl_xor(s, 1);
        s += __shfl_xor(s, 2);
        s += __shfl_xor(s, 4);
        s += __shfl_xor(s, 8);
        s += __shfl_xor(s, 16);
        if (lane == 0) sm.SumGlsPart[wave] = s;
    }
    for (int i = tid; i < H; i += 256) { sm.B1[i] = b1[i]; sm.B2[i] = b2[i]; }

    // vectorized transposing staging: coalesced float4 loads, scatter 4 LDS rows
    for (int v = tid; v < (H * D1) / 4; v += 256) {     // W1t[n][k] = W1[k][n]
        int k = v >> 6, n4 = v & 63;
        float4 w = ((const float4*)(W1 + k * H))[n4];
        sm.W1t[(n4 * 4 + 0) * W1S + k] = (_Float16)w.x;
        sm.W1t[(n4 * 4 + 1) * W1S + k] = (_Float16)w.y;
        sm.W1t[(n4 * 4 + 2) * W1S + k] = (_Float16)w.z;
        sm.W1t[(n4 * 4 + 3) * W1S + k] = (_Float16)w.w;
    }
    for (int v = tid; v < (H * H) / 4; v += 256) {      // W2t[n][k] = W2[k][n]
        int k = v >> 6, n4 = v & 63;
        float4 w = ((const float4*)(W2 + k * H))[n4];
        sm.W2t[(n4 * 4 + 0) * WS + k] = (_Float16)w.x;
        sm.W2t[(n4 * 4 + 1) * WS + k] = (_Float16)w.y;
        sm.W2t[(n4 * 4 + 2) * WS + k] = (_Float16)w.z;
        sm.W2t[(n4 * 4 + 3) * WS + k] = (_Float16)w.w;
    }
    for (int v = tid; v < (D * H) / 4; v += 256) {      // W3t[n][k] = W3[k][n]
        int k = v >> 4, n4 = v & 15;
        float4 w = ((const float4*)(W3 + k * D))[n4];
        sm.W3t[(n4 * 4 + 0) * WS + k] = (_Float16)w.x;
        sm.W3t[(n4 * 4 + 1) * WS + k] = (_Float16)w.y;
        sm.W3t[(n4 * 4 + 2) * WS + k] = (_Float16)w.z;
        sm.W3t[(n4 * 4 + 3) * WS + k] = (_Float16)w.w;
    }
    __syncthreads();

    _Float16* h1p = sm.scrA[wave].h;
    _Float16* h2p = sm.h2[wave];
    float*    yS  = sm.scrA[wave].y;               // aliases h1 (dead after L2)
    const float sumGls = sm.SumGlsPart[0] + sm.SumGlsPart[1];

    // ---------------- persistent loop over row tiles --------------------------
    for (int t = 0; t < TILES_PER_BLOCK; ++t) {
        const int rowBase = (blockIdx.x * TILES_PER_BLOCK + t) * ROWS_PER_BLOCK
                          + wave * ROWS_PER_WAVE;

        if (lane < ROWS_PER_WAVE)
            __builtin_prefetch(&x[(size_t)(rowBase + lane) * D], 0, 0);

        // ---- layer 1: h1 = gelu(x1 @ W1 + b1), K = 32 (perm gather fused) ----
        v16h a1;
        {
            const size_t xrow = (size_t)(rowBase + mrow) * D;
            const int koff = hiHalf * 8;
            #pragma unroll
            for (int i = 0; i < 8; ++i) {
                a1[i]     = (_Float16)x[xrow + (size_t)sm.InvPerm[koff + i]];
                a1[8 + i] = (_Float16)x[xrow + (size_t)sm.InvPerm[16 + koff + i]];
            }
        }
        #pragma unroll
        for (int nt = 0; nt < 16; ++nt) {
            const int n  = nt * 16 + mrow;
            const int kb = hiHalf * 8;
            v8h blo = *(const v8h*)&sm.W1t[n * W1S + kb];
            v8h bhi = *(const v8h*)&sm.W1t[n * W1S + kb + 16];
            v16h bf = join8(blo, bhi);
            v8f  c  = bcast8(sm.B1[n]);
            c = __builtin_amdgcn_wmma_f32_16x16x32_f16(false, a1, false, bf,
                                                       (short)0, c, false, false);
            #pragma unroll
            for (int r = 0; r < 8; ++r) {
                int m = r + hiHalf * 8;
                h1p[m * WS + n] = (_Float16)gelu_fast(c[r]);
            }
        }

        // ---- layer 2: h2 = gelu(h1 @ W2 + b2), K = 256 ----
        v8f acc2[16];
        #pragma unroll
        for (int nt = 0; nt < 16; ++nt)
            acc2[nt] = bcast8(sm.B2[nt * 16 + mrow]);

        #pragma unroll
        for (int kt = 0; kt < 8; ++kt) {
            const int ka = kt * 32 + hiHalf * 8;
            v8h lo = *(const v8h*)&h1p[mrow * WS + ka];
            v8h hi = *(const v8h*)&h1p[mrow * WS + ka + 16];
            v16h af = join8(lo, hi);
            const int kb = kt * 32 + hiHalf * 8;
            #pragma unroll
            for (int nt = 0; nt < 16; ++nt) {
                const int n = nt * 16 + mrow;
                v8h blo = *(const v8h*)&sm.W2t[n * WS + kb];
                v8h bhi = *(const v8h*)&sm.W2t[n * WS + kb + 16];
                v16h bf = join8(blo, bhi);
                acc2[nt] = __builtin_amdgcn_wmma_f32_16x16x32_f16(false, af, false, bf,
                                                                  (short)0, acc2[nt], false, false);
            }
        }
        #pragma unroll
        for (int nt = 0; nt < 16; ++nt) {
            const int n = nt * 16 + mrow;
            #pragma unroll
            for (int r = 0; r < 8; ++r) {
                int m = r + hiHalf * 8;
                h2p[m * WS + n] = (_Float16)gelu_fast(acc2[nt][r]);
            }
        }

        // ---- layer 3: params = h2 @ W3 + b3, N = 64 ----
        v8f acc3[4];
        #pragma unroll
        for (int nt = 0; nt < 4; ++nt)
            acc3[nt] = bcast8(sm.B3[nt * 16 + mrow]);

        #pragma unroll
        for (int kt = 0; kt < 8; ++kt) {
            const int ka = kt * 32 + hiHalf * 8;
            v8h lo = *(const v8h*)&h2p[mrow * WS + ka];
            v8h hi = *(const v8h*)&h2p[mrow * WS + ka + 16];
            v16h af = join8(lo, hi);
            const int kb = kt * 32 + hiHalf * 8;
            #pragma unroll
            for (int nt = 0; nt < 4; ++nt) {
                const int n = nt * 16 + mrow;
                v8h blo = *(const v8h*)&sm.W3t[n * WS + kb];
                v8h bhi = *(const v8h*)&sm.W3t[n * WS + kb + 16];
                v16h bf = join8(blo, bhi);
                acc3[nt] = __builtin_amdgcn_wmma_f32_16x16x32_f16(false, af, false, bf,
                                                                  (short)0, acc3[nt], false, false);
            }
        }

        // ---- tail: affine coupling + global scale + permutation scatter ----
        // x1 half: y[:, c<32] = x1 * egls + goff -> output column inv_perm[c]
        #pragma unroll
        for (int i = 0; i < 16; ++i) {
            int idx = i * 32 + lane;                // 16 rows x 32 cols
            int m = idx >> 5;
            int c = idx & 31;
            int ip = sm.InvPerm[c];
            float v = x[(size_t)(rowBase + m) * D + ip];
            yS[m * YS + ip] = __builtin_fmaf(v, sm.Egls[c], sm.Goff[c]);
        }

        // x2 half + log-Jacobian
        int   ipx2[2];
        float eg[2], go[2];
        #pragma unroll
        for (int tt = 0; tt < 2; ++tt) {
            int c    = 32 + 16 * tt + mrow;
            ipx2[tt] = sm.InvPerm[c];
            eg[tt]   = sm.Egls[c];
            go[tt]   = sm.Goff[c];
        }
        #pragma unroll
        for (int r = 0; r < 8; ++r) {
            int m = r + hiHalf * 8;
            const size_t xrow = (size_t)(rowBase + m) * D;
            float rsum = 0.0f;
            #pragma unroll
            for (int tt = 0; tt < 2; ++tt) {
                float ls  = 2.0f * tanh_fast(0.1f * acc3[tt][r]);  // tiles 0,1: log_scale
                float off = acc3[tt + 2][r];                       // tiles 2,3: offset
                rsum += ls;
                float x2v = x[xrow + ipx2[tt]];
                float y2  = __builtin_fmaf(x2v, __expf(ls), off);
                yS[m * YS + ipx2[tt]] = __builtin_fmaf(y2, eg[tt], go[tt]);
            }
            rsum += __shfl_xor(rsum, 1, 16);
            rsum += __shfl_xor(rsum, 2, 16);
            rsum += __shfl_xor(rsum, 4, 16);
            rsum += __shfl_xor(rsum, 8, 16);
            if ((lane & 15) == 0)
                out_lj[rowBase + m] = rsum + sumGls;
        }

        // coalesced copy-out: 16 rows x 64 f32 (permutation already in LDS)
        #pragma unroll
        for (int i = 0; i < 8; ++i) {
            int idx = i * 32 + lane;                // 0..255
            int m  = idx >> 4;
            int q4 = idx & 15;
            float4 v = ((const float4*)(yS + m * YS))[q4];
            ((float4*)(out_y + (size_t)(rowBase + m) * D))[q4] = v;
        }
    }
}

extern "C" void kernel_launch(void* const* d_in, const int* in_sizes, int n_in,
                              void* d_out, int out_size, void* d_ws, size_t ws_size,
                              hipStream_t stream) {
    (void)in_sizes; (void)n_in; (void)out_size; (void)d_ws; (void)ws_size;
    const float* x    = (const float*)d_in[0];
    const int*   perm = (const int*)  d_in[1];
    const float* W1   = (const float*)d_in[2];
    const float* b1   = (const float*)d_in[3];
    const float* W2   = (const float*)d_in[4];
    const float* b2   = (const float*)d_in[5];
    const float* W3   = (const float*)d_in[6];
    const float* b3   = (const float*)d_in[7];
    const float* gls  = (const float*)d_in[8];
    const float* gof  = (const float*)d_in[9];

    float* out_y  = (float*)d_out;
    float* out_lj = out_y + (size_t)B_TOTAL * D;

    dim3 grid(NBLOCKS);   // 512 persistent blocks x 2 tiles
    flow_fused_kernel<<<grid, 256, 0, stream>>>(x, perm, W1, b1, W2, b2, W3, b3,
                                                gls, gof, out_y, out_lj);
}
